// RwkvSelfAttention_31507880083539
// MI455X (gfx1250) — matrix operand order repaired and checked by reading, not compiled, not run on hardware
//
#include <hip/hip_runtime.h>
#include <stdint.h>

// ---------------------------------------------------------------------------
// RWKV v4 self-attention for MI455X (gfx1250, wave32, WMMA).
//   B=4, S=2048, H=D=2048  (B hardcoded; S derived from in_sizes).
// GEMMs in bf16 via v_wmma_f32_16x16x32_bf16 (f32 accumulate) with
// double-buffered LDS fed by GLOBAL_LOAD_ASYNC_TO_LDS_B128 (ASYNCcnt),
// software-pipelined one K-tile ahead; steady-state loop is branch-free.
// Scan in f32. Workspace ~370 MB.
// ---------------------------------------------------------------------------

typedef __attribute__((ext_vector_type(16))) __bf16          v16bf;
typedef __attribute__((ext_vector_type(8)))  float           v8f;
typedef __attribute__((ext_vector_type(8)))  unsigned short  ushort8;

union Frag16 { v16bf v; ushort8 h[2]; };

__device__ __forceinline__ unsigned short f2bf(float f) {
  union { float f; unsigned u; } x; x.f = f;
  unsigned r = x.u + 0x7FFFu + ((x.u >> 16) & 1u);   // round-to-nearest-even
  return (unsigned short)(r >> 16);
}

// Async global -> LDS copy, 16 bytes per lane. GVS addressing:
// uniform 64-bit SGPR base + per-lane 32-bit byte offset. Tracked by ASYNCcnt.
__device__ __forceinline__ void async_copy_b128(unsigned lds_off, unsigned voff,
                                                const void* sbase) {
  asm volatile("global_load_async_to_lds_b128 %0, %1, %2"
               :: "v"(lds_off), "v"(voff), "s"(sbase)
               : "memory");
}

// ------------------------- weight f32 -> bf16 ------------------------------
__global__ void cvt_f32_bf16(const float* __restrict__ in,
                             unsigned short* __restrict__ out, int n) {
  int i = (blockIdx.x * blockDim.x + threadIdx.x) * 4;
  if (i >= n) return;
  float4 v = *(const float4*)(in + i);
  uint2 o;
  o.x = (unsigned)f2bf(v.x) | ((unsigned)f2bf(v.y) << 16);
  o.y = (unsigned)f2bf(v.z) | ((unsigned)f2bf(v.w) << 16);
  *(uint2*)(out + i) = o;
}

// ---------------- time-shift + 3 mixes, fused, f32 -> bf16 -----------------
__device__ __forceinline__ uint2 mix_pack(float4 h, float4 sh, float4 m) {
  float a = h.x * m.x + sh.x * (1.f - m.x);
  float b = h.y * m.y + sh.y * (1.f - m.y);
  float c = h.z * m.z + sh.z * (1.f - m.z);
  float d = h.w * m.w + sh.w * (1.f - m.w);
  uint2 o;
  o.x = (unsigned)f2bf(a) | ((unsigned)f2bf(b) << 16);
  o.y = (unsigned)f2bf(c) | ((unsigned)f2bf(d) << 16);
  return o;
}

__global__ void rwkv_mix(const float* __restrict__ h,
                         const float* __restrict__ mk,
                         const float* __restrict__ mv,
                         const float* __restrict__ mr,
                         unsigned short* __restrict__ xk,
                         unsigned short* __restrict__ xv,
                         unsigned short* __restrict__ xr,
                         int S, int H, size_t total) {
  size_t i = ((size_t)blockIdx.x * blockDim.x + threadIdx.x) * 4;
  if (i >= total) return;
  float4 hv = *(const float4*)(h + i);
  size_t row = i / (size_t)H;
  int s = (int)(row % (size_t)S);
  float4 sh = make_float4(0.f, 0.f, 0.f, 0.f);
  if (s != 0) sh = *(const float4*)(h + i - (size_t)H);
  int hp = (int)(i % (size_t)H);
  float4 k4 = *(const float4*)(mk + hp);
  float4 v4 = *(const float4*)(mv + hp);
  float4 r4 = *(const float4*)(mr + hp);
  *(uint2*)(xk + i) = mix_pack(hv, sh, k4);
  *(uint2*)(xv + i) = mix_pack(hv, sh, v4);
  *(uint2*)(xr + i) = mix_pack(hv, sh, r4);
}

// --------------------------- bf16 WMMA GEMM --------------------------------
// C[M,N] = A[M,K] * W[N,K]^T   (torch Linear layout: W is [out,in], so a
// B-matrix column n == contiguous row n of W — no transpose needed).
// Block: 256 threads = 8 waves; tile BM=128 x BN=128, BK=32, double-buffered
// LDS fed by async-to-LDS loads one K-tile ahead (last tile peeled).
// Each wave: 32x64 patch = 2x4 tiles of 16x16, accum in 64 VGPRs; A-operand
// reuse hints set (literal true) on the first 3 of each group of 4 identical
// WMMAs sharing SRC0.
// mode: 0 = plain store, 1 = sigmoid epilogue.
__global__ __launch_bounds__(256, 2) void gemm_bf16_wmma(
    const unsigned short* __restrict__ A,
    const unsigned short* __restrict__ W,
    float* __restrict__ C,
    int M, int N, int K, int mode) {
  constexpr int BM = 128, BN = 128, BK = 32, LDSK = 40;  // pad 32->40 elems
  __shared__ unsigned short As[2][BM * LDSK];
  __shared__ unsigned short Bs[2][BN * LDSK];

  const int tid  = threadIdx.x;
  const int lane = tid & 31;
  const int wave = tid >> 5;
  const int wm   = wave & 3;   // 4 waves along M, 32 rows each
  const int wn   = wave >> 2;  // 2 waves along N, 64 cols each
  const int half = lane >> 4;  // K-half selector per ISA wave32 layout
  const int l16  = lane & 15;
  const unsigned bm = blockIdx.y * BM;
  const unsigned bn = blockIdx.x * BN;

  // LDS byte addresses of the two buffers (low 32 bits of flat LDS address).
  const unsigned asb[2] = { (unsigned)(uintptr_t)&As[0][0],
                            (unsigned)(uintptr_t)&As[1][0] };
  const unsigned bsb[2] = { (unsigned)(uintptr_t)&Bs[0][0],
                            (unsigned)(uintptr_t)&Bs[1][0] };

  // Loop-invariant per-thread copy assignments: 2 x 16B chunks of A and of B.
  unsigned lofs[2], voffa[2], voffb[2];
#pragma unroll
  for (int i = 0; i < 2; ++i) {
    unsigned seg = (unsigned)tid + i * 256u;
    unsigned row = seg >> 2;
    unsigned col = (seg & 3u) << 3;
    lofs[i]  = (row * LDSK + col) * 2u;               // LDS byte offset
    voffa[i] = ((bm + row) * (unsigned)K + col) * 2u; // global byte offset
    voffb[i] = ((bn + row) * (unsigned)K + col) * 2u;
  }

  v8f acc[2][4];
  const v8f vz = {0.f, 0.f, 0.f, 0.f, 0.f, 0.f, 0.f, 0.f};
#pragma unroll
  for (int mi = 0; mi < 2; ++mi)
#pragma unroll
    for (int ni = 0; ni < 4; ++ni) acc[mi][ni] = vz;

  auto issue_tile = [&](int buf, int kt) {
    const void* abase = (const char*)A + (size_t)kt * 2;
    const void* bbase = (const char*)W + (size_t)kt * 2;
#pragma unroll
    for (int i = 0; i < 2; ++i) {
      async_copy_b128(asb[buf] + lofs[i], voffa[i], abase);
      async_copy_b128(bsb[buf] + lofs[i], voffb[i], bbase);
    }
  };

  auto compute_tile = [&](int cur) {
    const unsigned short* as = &As[cur][0];
    const unsigned short* bs = &Bs[cur][0];
    // A fragment (16-bit 16x32): lanes 0-15 row m hold K 0-7 & 16-23,
    // lanes 16-31 hold K 8-15 & 24-31.
    Frag16 af[2];
#pragma unroll
    for (int mi = 0; mi < 2; ++mi) {
      int row = wm * 32 + mi * 16 + l16;
      af[mi].h[0] = *(const ushort8*)&as[row * LDSK + half * 8];
      af[mi].h[1] = *(const ushort8*)&as[row * LDSK + half * 8 + 16];
    }
    // B fragment (32x16): lanes 0-15 col n hold K 0-7 & 16-23, lanes 16-31 K 8-15 & 24-31.
    Frag16 bfr[4];
#pragma unroll
    for (int ni = 0; ni < 4; ++ni) {
      int coln = wn * 64 + ni * 16 + l16;
      bfr[ni].h[0] = *(const ushort8*)&bs[coln * LDSK + half * 8];
      bfr[ni].h[1] = *(const ushort8*)&bs[coln * LDSK + half * 8 + 16];
    }
    // reuse flags must be literal constants for the builtin.
#define RWKV_WMMA(MI, NI, RA)                                                \
    acc[MI][NI] = __builtin_amdgcn_wmma_f32_16x16x32_bf16(                   \
        false, af[MI].v, false, bfr[NI].v, (short)0, acc[MI][NI], RA, false)
    RWKV_WMMA(0, 0, true);  RWKV_WMMA(0, 1, true);
    RWKV_WMMA(0, 2, true);  RWKV_WMMA(0, 3, false);
    RWKV_WMMA(1, 0, true);  RWKV_WMMA(1, 1, true);
    RWKV_WMMA(1, 2, true);  RWKV_WMMA(1, 3, false);
#undef RWKV_WMMA
  };

  // Prime buffer 0 with the first K-tile.
  issue_tile(0, 0);

  // Steady state: branch-free — issue next tile, wait current, compute.
  int kt = 0;
  for (; kt < K - BK; kt += BK) {
    const int cur = (kt >> 5) & 1;
    issue_tile(cur ^ 1, kt + BK);
    asm volatile("s_wait_asynccnt 0x4" ::: "memory");  // current tile landed
    __syncthreads();
    compute_tile(cur);
    __syncthreads();  // done reading buf[cur]; it is refilled next iteration
  }
  // Peeled last tile: nothing left to prefetch.
  {
    const int cur = (kt >> 5) & 1;
    asm volatile("s_wait_asynccnt 0x0" ::: "memory");
    __syncthreads();
    compute_tile(cur);
  }

  // Epilogue: C/D 16x16 f32 layout — VGPR i, lane l: M = i + (l>=16)*8, N = l%16.
#pragma unroll
  for (int mi = 0; mi < 2; ++mi) {
    size_t row0 = (size_t)bm + wm * 32 + mi * 16 + half * 8;
#pragma unroll
    for (int ni = 0; ni < 4; ++ni) {
      size_t col = (size_t)bn + wn * 64 + ni * 16 + l16;
#pragma unroll
      for (int i = 0; i < 8; ++i) {
        float x = acc[mi][ni][i];
        if (mode == 1) x = 1.0f / (1.0f + __expf(-x));
        C[(row0 + i) * (size_t)N + col] = x;
      }
    }
  }
}

// -------------------- WKV recurrence (sequential over S) -------------------
// One lane per (b,d) channel; fuses r * wkv and converts to bf16 for the
// final GEMM. Stable log-sum-exp form, matching the reference exactly.
__global__ void wkv_scan(const float* __restrict__ Kf,
                         const float* __restrict__ Vf,
                         const float* __restrict__ Rf,
                         const float* __restrict__ td,
                         const float* __restrict__ tf,
                         unsigned short* __restrict__ RR,
                         int S, int D) {
  int d = blockIdx.x * blockDim.x + threadIdx.x;
  int b = blockIdx.y;
  float w = -__expf(td[d]);
  float u = tf[d];
  float num = 0.f, den = 0.f, mx = -1e38f;
  size_t base = ((size_t)b * S) * (size_t)D + d;
  size_t stride = (size_t)D;
  for (int s = 0; s < S; ++s) {
    size_t idx = base + (size_t)s * stride;
    if (s + 4 < S) {
      __builtin_prefetch(Kf + idx + 4 * stride, 0, 1);
      __builtin_prefetch(Vf + idx + 4 * stride, 0, 1);
      __builtin_prefetch(Rf + idx + 4 * stride, 0, 1);
    }
    float k = Kf[idx];
    float v = Vf[idx];
    // output for current token (boosted by time_first)
    float mo = fmaxf(mx, k + u);
    float e1 = __expf(mx - mo);
    float e2 = __expf(k + u - mo);
    float out = (e1 * num + e2 * v) / (e1 * den + e2);
    // state update with decay
    float mn = fmaxf(mx + w, k);
    e1 = __expf(mx + w - mn);
    e2 = __expf(k - mn);
    num = e1 * num + e2 * v;
    den = e1 * den + e2;
    mx = mn;
    float r = Rf[idx];  // already sigmoid'ed in GEMM epilogue
    RR[idx] = f2bf(r * out);
  }
}

// ---------------------------------------------------------------------------
extern "C" void kernel_launch(void* const* d_in, const int* in_sizes, int n_in,
                              void* d_out, int out_size, void* d_ws,
                              size_t ws_size, hipStream_t stream) {
  const float* hidden = (const float*)d_in[0];
  const float* td     = (const float*)d_in[1];
  const float* tf     = (const float*)d_in[2];
  const float* mk     = (const float*)d_in[3];
  const float* mv     = (const float*)d_in[4];
  const float* mr     = (const float*)d_in[5];
  const float* Wk     = (const float*)d_in[6];
  const float* Wv     = (const float*)d_in[7];
  const float* Wr     = (const float*)d_in[8];
  const float* Wo     = (const float*)d_in[9];
  float* out          = (float*)d_out;

  const int H = in_sizes[1];                       // D == H == 2048
  const int B = 4;                                 // fixed by the reference
  const int S = (int)(((size_t)in_sizes[0] / H) / B);
  const int M = B * S;                             // 8192
  const int N = H, K = H;
  const size_t MH = (size_t)M * H;
  const size_t HH = (size_t)H * H;

  // workspace carve-up
  char* p = (char*)d_ws;
  unsigned short* xk  = (unsigned short*)p; p += MH * 2;
  unsigned short* xv  = (unsigned short*)p; p += MH * 2;
  unsigned short* xr  = (unsigned short*)p; p += MH * 2;
  unsigned short* wkb = (unsigned short*)p; p += HH * 2;
  unsigned short* wvb = (unsigned short*)p; p += HH * 2;
  unsigned short* wrb = (unsigned short*)p; p += HH * 2;
  unsigned short* wob = (unsigned short*)p; p += HH * 2;
  float* kf = (float*)p; p += MH * 4;
  float* vf = (float*)p; p += MH * 4;
  float* rf = (float*)p; p += MH * 4;
  unsigned short* rr = (unsigned short*)p; p += MH * 2;

  // 1) weights -> bf16
  {
    int n = (int)HH;
    int grid = (n / 4 + 255) / 256;
    cvt_f32_bf16<<<grid, 256, 0, stream>>>(Wk, wkb, n);
    cvt_f32_bf16<<<grid, 256, 0, stream>>>(Wv, wvb, n);
    cvt_f32_bf16<<<grid, 256, 0, stream>>>(Wr, wrb, n);
    cvt_f32_bf16<<<grid, 256, 0, stream>>>(Wo, wob, n);
  }

  // 2) fused time-shift + mixes -> bf16
  {
    size_t nq = MH / 4;
    int grid = (int)((nq + 255) / 256);
    rwkv_mix<<<grid, 256, 0, stream>>>(hidden, mk, mv, mr, xk, xv, xr, S, H, MH);
  }

  // 3) k / v / r GEMMs (sigmoid fused into r)
  dim3 ggrid(N / 128, M / 128);
  gemm_bf16_wmma<<<ggrid, 256, 0, stream>>>(xk, wkb, kf, M, N, K, 0);
  gemm_bf16_wmma<<<ggrid, 256, 0, stream>>>(xv, wvb, vf, M, N, K, 0);
  gemm_bf16_wmma<<<ggrid, 256, 0, stream>>>(xr, wrb, rf, M, N, K, 1);

  // 4) WKV recurrence, fused with r*wkv, emits bf16
  wkv_scan<<<dim3(H / 256, B), 256, 0, stream>>>(kf, vf, rf, td, tf, rr, S, H);

  // 5) output GEMM -> f32 d_out
  gemm_bf16_wmma<<<ggrid, 256, 0, stream>>>(rr, wob, out, M, N, K, 0);
}
